// STGCN_70317204570427
// MI455X (gfx1250) — compile-verified, matrix operand order checked by reading, not run on hardware
//
#include <hip/hip_runtime.h>

// ---------------------------------------------------------------------------
// CDNA5 (gfx1250) ST-GCN forward: bf16 WMMA pipeline, f32 accumulate.
// Activations between blocks: (N, T, V, C) bf16 -> channel-contiguous, so all
// tile staging is aligned b128. Temporal conv uses GLOBAL_LOAD_ASYNC_TO_LDS
// (ASYNCcnt-tracked DMA) with double-buffered LDS tiles.
// ---------------------------------------------------------------------------

typedef __attribute__((ext_vector_type(16))) __bf16 v16bf;
typedef __attribute__((ext_vector_type(8)))  float  v8f;

#define LDST 40   // u16 per LDS tile row: byte stride 80 (b128-aligned cols)

__device__ __forceinline__ unsigned short f2bf(float f) {
  union { float f; unsigned int u; } x; x.f = f;
  unsigned int r = x.u + 0x7FFFu + ((x.u >> 16) & 1u);   // round-to-nearest-even
  return (unsigned short)(r >> 16);
}

__device__ __forceinline__ unsigned int pack2(float a, float b) {
  return (unsigned int)f2bf(a) | ((unsigned int)f2bf(b) << 16);
}

// Async DMA: 16 bytes global -> LDS, per active lane. Tracked by ASYNCcnt.
// VDST operand carries the 32-bit LDS byte address (low 32 bits of a generic
// pointer to LDS are the LDS offset per the flat-aperture mapping).
__device__ __forceinline__ void async_g2l_b128(unsigned short* lds,
                                               const unsigned short* gsrc) {
  unsigned lds32 = (unsigned)(unsigned long long)lds;
  asm volatile("global_load_async_to_lds_b128 %0, %1, off"
               :: "v"(lds32), "v"(gsrc) : "memory");
}

// Load a 16x32 bf16 fragment (A layout; B uses the same via K-major-per-column
// LDS tiles).  ISA 7.12.2 (16-bit A-matrix 16x32):
//   lanes 0-15 : M=lane,    VGPR0-3: K={0,1..6,7},  VGPR4-7: K=16..23
//   lanes 16-31: M=lane-16, K offset +8 (and 24..31 in VGPR4-7)
__device__ __forceinline__ v16bf load_frag(const unsigned short* tile, int lane) {
  union { v16bf v; unsigned int u[8]; } f;
  int m   = lane & 15;
  int khi = (lane >> 4) << 3;                       // 0 or 8
  const unsigned int* row = (const unsigned int*)(tile + m * LDST);
  #pragma unroll
  for (int r = 0; r < 8; ++r) {
    int k = ((r & 4) << 2) + khi + ((r & 3) << 1);  // (r>=4?16:0)+khi+(r&3)*2
    f.u[r] = row[k >> 1];                           // bf16 pair (k, k+1)
  }
  return f.v;
}

// ---------------------------------------------------------------------------
// Pack kernels (fp32 -> padded bf16)
// ---------------------------------------------------------------------------

__global__ void k_pack_x(const float* __restrict__ x, unsigned short* __restrict__ f0,
                         long total) {
  long i = (long)blockIdx.x * blockDim.x + threadIdx.x;
  if (i < total) f0[i] = f2bf(x[i]);
}

__global__ void k_pack_ws(const float* __restrict__ ws, unsigned short* __restrict__ wsp,
                          int Cout, int Cin, int Cinp) {
  int i = blockIdx.x * blockDim.x + threadIdx.x;
  if (i >= Cout * Cinp) return;
  int ci = i % Cinp, o = i / Cinp;
  wsp[i] = (ci < Cin) ? f2bf(ws[(long)o * Cin + ci]) : (unsigned short)0;
}

// wt[o][i][k] (Cout x Cin x 9) -> wtp[k][o][ip] (9 x Cout x Cinp)
__global__ void k_pack_wt(const float* __restrict__ wt, unsigned short* __restrict__ wtp,
                          int Cout, int Cin, int Cinp) {
  long idx = (long)blockIdx.x * blockDim.x + threadIdx.x;
  long total = (long)9 * Cout * Cinp;
  if (idx >= total) return;
  int ip = (int)(idx % Cinp); long r = idx / Cinp;
  int o = (int)(r % Cout); int k = (int)(r / Cout);
  wtp[idx] = (ip < Cin) ? f2bf(wt[((long)o * Cin + ip) * 9 + k]) : (unsigned short)0;
}

__global__ void k_pack_adj(const float* __restrict__ A, unsigned short* __restrict__ adjp) {
  int i = blockIdx.x * blockDim.x + threadIdx.x;
  if (i >= 32 * 32) return;
  int w = i & 31, v = i >> 5;
  adjp[i] = (v < 25 && w < 25) ? f2bf(A[v * 25 + w]) : (unsigned short)0;
}

// ---------------------------------------------------------------------------
// Fused spatial kernel: G[n,t,w,o] = relu( sum_{c,v} Ws[o,c]*F[n,t,v,c]*A[v,w] )
// One workgroup (8 waves) per (n,t); waves own 16-row o-subtiles.
// ---------------------------------------------------------------------------
__global__ void __launch_bounds__(256)
k_spatial(const unsigned short* __restrict__ fin,   // (N, T, 25, Cin) bf16
          const unsigned short* __restrict__ wsp,   // (Cout, Cinp)    bf16
          const unsigned short* __restrict__ adjp,  // (32, 32) [v][w] bf16
          unsigned short* __restrict__ gout,        // (N, T, 25, Cout) bf16
          int Cin, int Cinp, int Cout, int T) {
  __shared__ __align__(16) unsigned short Wss[256 * LDST];  // A tile: Cout x 32
  __shared__ __align__(16) unsigned short Fs [32  * LDST];  // B tile: [v][k]
  __shared__ __align__(16) unsigned short Adjs[32 * LDST];  // B tile: [w][v]
  __shared__ __align__(16) unsigned short Hs [128 * LDST];  // per-wave 16x32 H

  const int n = blockIdx.y, t = blockIdx.x;
  const int tid = threadIdx.x, lane = tid & 31, wave = tid >> 5;
  const int numO16 = Cout >> 4;
  const int maxG = (numO16 + 7) >> 3;     // 1 (Cout<=128) or 2 (Cout==256)

  for (int idx = tid; idx < 1024; idx += 256) {    // Adj -> LDS, transposed
    int w = idx & 31, v = idx >> 5;
    Adjs[w * LDST + v] = adjp[idx];
  }

  const v8f vzero = {0.f,0.f,0.f,0.f,0.f,0.f,0.f,0.f};
  v8f acc[2][2];
  acc[0][0] = vzero; acc[0][1] = vzero; acc[1][0] = vzero; acc[1][1] = vzero;

  const unsigned short* finBase = fin + (((long)n * T + t) * 25) * Cin; // [v][c]

  // ----- Stage 1: H = Ws * F -----
  for (int kb = 0; kb < Cinp; kb += 32) {
    int nvec = Cout << 2;                        // Cout*32/8 uint4 slots
    for (int e = tid; e < nvec; e += 256) {
      int elem = e << 3;
      int o = elem >> 5, kk = elem & 31;
      *(uint4*)(Wss + o * LDST + kk) =
          *(const uint4*)(wsp + (long)o * Cinp + kb + kk);
    }
    if (Cin >= 32) {
      if (tid < 128) {
        int elem = tid << 3;
        int v = elem >> 5, kk = elem & 31;
        uint4 val = {0u,0u,0u,0u};
        if (v < 25) val = *(const uint4*)(finBase + v * Cin + kb + kk);
        *(uint4*)(Fs + v * LDST + kk) = val;
        if (v < 25 && kb + 32 < Cinp)            // emits global_prefetch_b8
          __builtin_prefetch(finBase + v * Cin + kb + 32 + kk, 0, 0);
      }
    } else {
      for (int idx = tid; idx < 1024; idx += 256) {
        int kk = idx & 31, v = idx >> 5;
        int c = kb + kk;
        unsigned short val = 0;
        if (v < 25 && c < Cin) val = finBase[v * Cin + c];
        Fs[v * LDST + kk] = val;
      }
    }
    __syncthreads();
    #pragma unroll
    for (int g = 0; g < 2; ++g) {
      if (g >= maxG) break;
      int og = wave + 8 * g;
      if (og >= numO16) og = 0;                // clamp: uniform, branch-free WMMA
      v16bf a  = load_frag(Wss + og * 16 * LDST, lane);
      v16bf b0 = load_frag(Fs, lane);
      v16bf b1 = load_frag(Fs + 16 * LDST, lane);
      acc[g][0] = __builtin_amdgcn_wmma_f32_16x16x32_bf16(
          false, a, false, b0, (short)0, acc[g][0], false, false);
      acc[g][1] = __builtin_amdgcn_wmma_f32_16x16x32_bf16(
          false, a, false, b1, (short)0, acc[g][1], false, false);
    }
    __syncthreads();
  }

  // ----- Stage 2: G = relu(H * Adj), store channel-contiguous -----
  const long outBase = (((long)n * T + t) * 25) * Cout;   // + w*Cout + o
  #pragma unroll
  for (int g = 0; g < 2; ++g) {
    if (g >= maxG) break;
    int ogRaw = wave + 8 * g;
    bool active = (ogRaw < numO16);
    if (active) {
      int ncol = lane & 15, mh = (lane >> 4) << 3;
      unsigned short* hrow = Hs + wave * 16 * LDST;
      #pragma unroll
      for (int r = 0; r < 8; ++r) {
        hrow[(r + mh) * LDST + ncol]      = f2bf(acc[g][0][r]);
        hrow[(r + mh) * LDST + 16 + ncol] = f2bf(acc[g][1][r]);
      }
    }
    __syncthreads();
    if (active) {
      v16bf h  = load_frag(Hs + wave * 16 * LDST, lane);
      v16bf a0 = load_frag(Adjs, lane);
      v16bf a1 = load_frag(Adjs + 16 * LDST, lane);
      v8f d0 = __builtin_amdgcn_wmma_f32_16x16x32_bf16(
          false, h, false, a0, (short)0, vzero, false, false);
      v8f d1 = __builtin_amdgcn_wmma_f32_16x16x32_bf16(
          false, h, false, a1, (short)0, vzero, false, false);
      int wc = lane & 15, mh = (lane >> 4) << 3;
      int o0 = ogRaw * 16 + mh;
      #pragma unroll
      for (int r = 0; r < 8; r += 2) {
        *(unsigned int*)(gout + outBase + (long)wc * Cout + o0 + r) =
            pack2(fmaxf(d0[r], 0.f), fmaxf(d0[r + 1], 0.f));
        if (wc < 9)
          *(unsigned int*)(gout + outBase + (long)(16 + wc) * Cout + o0 + r) =
              pack2(fmaxf(d1[r], 0.f), fmaxf(d1[r + 1], 0.f));
      }
    }
    __syncthreads();
  }
}

// ---------------------------------------------------------------------------
// Temporal conv (gamma=9, pad 4, stride s) as implicit-im2col WMMA GEMM.
// Tile: 64 (o) x 64 (flat to*25+w). Double-buffered async-to-LDS staging:
// each iteration issues next slab's DMA, waits ASYNCcnt<=2, computes current.
// ---------------------------------------------------------------------------
__global__ void __launch_bounds__(256)
k_temporal(const unsigned short* __restrict__ gin,  // (N, Tin, 25, Ci) bf16
           const unsigned short* __restrict__ wtp,  // (9, Cout, Ci)    bf16
           unsigned short* __restrict__ outH,       // (N, Tout, 25, Cout) bf16
           float* __restrict__ outF,                // (N, Cout, Tout, 25) f32
           int useF32, int Ci, int Tin, int Tout, int s, int Cout) {
  __shared__ __align__(16) unsigned short Wts[2][64 * LDST];  // A: 64 o x 32 i
  __shared__ __align__(16) unsigned short Gs [2][64 * LDST];  // B: [col][i]

  const int n = blockIdx.z;
  const int oBase = blockIdx.y * 64;
  const int colBase = blockIdx.x * 64;
  const int tid = threadIdx.x, lane = tid & 31, wave = tid >> 5;
  const int osub = wave >> 1, cpair = wave & 1;
  const int ncolsTot = Tout * 25;

  // Fixed per-thread staging slots (one b128 each): loop-invariant mapping.
  const int oA = tid >> 2, kA = (tid & 3) << 3;     // A: row o, K offset
  const int jB = oA, kB = kA;                       // B: col j, K offset
  const int jgB = colBase + jB;
  const int toB = jgB / 25;
  const int wB  = jgB - toB * 25;
  const bool colOK = (jgB < ncolsTot);
  const long wtRow = (long)(oBase + oA) * Ci + kA;  // includes kA
  const long CoutCi = (long)Cout * Ci;

  const v8f vzero = {0.f,0.f,0.f,0.f,0.f,0.f,0.f,0.f};
  v8f acc0 = vzero, acc1 = vzero;

  const int KI = Ci >> 5;
  const int itTot = 9 * KI;

  // Prologue: issue tile 0 into buffer 0 (always 2 async ops per thread).
  {
    async_g2l_b128(&Wts[0][oA * LDST + kA], wtp + wtRow);
    int tt = toB * s - 4;                            // k=0
    const unsigned short* src = (colOK && tt >= 0 && tt < Tin)
        ? gin + (((long)n * Tin + tt) * 25 + wB) * Ci + kB
        : gin;                                       // clamped-safe dummy
    async_g2l_b128(&Gs[0][jB * LDST + kB], src);
  }

  int k = 0, kb = 0;
  for (int it = 0; it < itTot; ++it) {
    const int cur = it & 1;
    int kb2 = kb + 32, k2 = k;
    if (kb2 == Ci) { kb2 = 0; ++k2; }

    if (it + 1 < itTot) {
      const int nxt = cur ^ 1;
      async_g2l_b128(&Wts[nxt][oA * LDST + kA],
                     wtp + (long)k2 * CoutCi + wtRow + kb2);
      int tt2 = toB * s + (k2 - 4);
      const unsigned short* src2 = (colOK && tt2 >= 0 && tt2 < Tin)
          ? gin + (((long)n * Tin + tt2) * 25 + wB) * Ci + kb2 + kB
          : gin;
      async_g2l_b128(&Gs[nxt][jB * LDST + kB], src2);
      asm volatile("s_wait_asynccnt 2" ::: "memory");  // tile 'it' DMA done
    } else {
      asm volatile("s_wait_asynccnt 0" ::: "memory");
    }

    // Patch out-of-range columns of the *current* tile with zeros (own DMA
    // for this slot already completed; others read only after the barrier).
    int tt = toB * s + (k - 4);
    if (!(colOK && tt >= 0 && tt < Tin)) {
      uint4 z = {0u,0u,0u,0u};
      *(uint4*)(&Gs[cur][jB * LDST + kB]) = z;
    }
    __syncthreads();

    v16bf a  = load_frag(&Wts[cur][osub * 16 * LDST], lane);
    v16bf b0 = load_frag(&Gs [cur][(cpair * 32) * LDST], lane);
    v16bf b1 = load_frag(&Gs [cur][(cpair * 32 + 16) * LDST], lane);
    acc0 = __builtin_amdgcn_wmma_f32_16x16x32_bf16(
        false, a, false, b0, (short)0, acc0, false, false);
    acc1 = __builtin_amdgcn_wmma_f32_16x16x32_bf16(
        false, a, false, b1, (short)0, acc1, false, false);
    __syncthreads();

    kb = kb2; k = k2;
  }

  // ---- Epilogue: ReLU + store ----
  const int mh = (lane >> 4) << 3;
  #pragma unroll
  for (int c2 = 0; c2 < 2; ++c2) {
    v8f d = (c2 == 0) ? acc0 : acc1;
    int jg = colBase + cpair * 32 + c2 * 16 + (lane & 15);
    if (jg < ncolsTot) {
      int to = jg / 25;
      int w  = jg - to * 25;
      if (useF32) {
        #pragma unroll
        for (int r = 0; r < 8; ++r) {
          int o = oBase + osub * 16 + r + mh;
          outF[(((long)n * Cout + o) * Tout + to) * 25 + w] = fmaxf(d[r], 0.f);
        }
      } else {
        long off = (((long)n * Tout + to) * 25 + w) * Cout + oBase + osub * 16 + mh;
        #pragma unroll
        for (int r = 0; r < 8; r += 2)
          *(unsigned int*)(outH + off + r) =
              pack2(fmaxf(d[r], 0.f), fmaxf(d[r + 1], 0.f));
      }
    }
  }
}

// ---------------------------------------------------------------------------
// Host launcher
// ---------------------------------------------------------------------------
struct BlkCfg { int cin, cout, s; };
static const BlkCfg g_blks[10] = {
    {3, 64, 1},  {64, 64, 1},  {64, 64, 1},  {64, 64, 1},
    {64, 128, 2},{128, 128, 1},{128, 128, 1},
    {128, 256, 2},{256, 256, 1},{256, 256, 1}};

extern "C" void kernel_launch(void* const* d_in, const int* in_sizes, int n_in,
                              void* d_out, int out_size, void* d_ws, size_t ws_size,
                              hipStream_t stream) {
  (void)in_sizes; (void)n_in; (void)out_size; (void)ws_size;
  const float* x = (const float*)d_in[0];
  const float* A = (const float*)d_in[1];

  const int N = 64, V = 25;
  int T = 300;

  // Workspace layout (u16 units, all region offsets 16B-aligned):
  unsigned short* base   = (unsigned short*)d_ws;
  unsigned short* actCur = base;                       // <= 30,720,000 elems
  unsigned short* actTmp = actCur + 30720000ul;        // <= 61,440,000 elems
  unsigned short* adjp   = actTmp + 61440000ul;        // 1024
  unsigned short* wpool  = adjp + 1024;                // packed weights

  k_pack_adj<<<4, 256, 0, stream>>>(A, adjp);
  {
    long tot = (long)N * 300 * V * 3;
    k_pack_x<<<(int)((tot + 255) / 256), 256, 0, stream>>>(x, actCur, tot);
  }

  unsigned short* wspP[10];
  unsigned short* wtpP[10];
  unsigned short* p = wpool;
  for (int i = 0; i < 10; ++i) {
    int cin = g_blks[i].cin, cout = g_blks[i].cout;
    int cinp = ((cin + 31) / 32) * 32;
    wspP[i] = p; p += (size_t)cout * cinp;
    wtpP[i] = p; p += (size_t)9 * cout * cout;

    const float* ws_i = (const float*)d_in[2 + 2 * i];
    const float* wt_i = (const float*)d_in[3 + 2 * i];
    {
      int tot = cout * cinp;
      k_pack_ws<<<(tot + 255) / 256, 256, 0, stream>>>(ws_i, wspP[i], cout, cin, cinp);
    }
    {
      long tot = (long)9 * cout * cout;
      k_pack_wt<<<(int)((tot + 255) / 256), 256, 0, stream>>>(wt_i, wtpP[i], cout, cout, cout);
    }
  }

  for (int i = 0; i < 10; ++i) {
    int cin = g_blks[i].cin, cout = g_blks[i].cout, s = g_blks[i].s;
    int cinp = ((cin + 31) / 32) * 32;
    int Tin = T;
    int Tout = (Tin + s - 1) / s;

    // Spatial: actCur (N,Tin,V,cin) -> actTmp (N,Tin,V,cout)
    dim3 gs(Tin, N);
    k_spatial<<<gs, 256, 0, stream>>>(actCur, wspP[i], adjp, actTmp,
                                      cin, cinp, cout, Tin);

    // Temporal: actTmp -> actCur (bf16) or d_out (f32 NCHW, last block)
    int ncols = Tout * V;
    dim3 gt((ncols + 63) / 64, cout / 64, N);
    int last = (i == 9);
    k_temporal<<<gt, 256, 0, stream>>>(actTmp, wtpP[i],
                                       last ? (unsigned short*)nullptr : actCur,
                                       last ? (float*)d_out : (float*)nullptr,
                                       last, cout, Tin, Tout, s, cout);
    T = Tout;
  }
}
